// AttnBlock_12326556139896
// MI455X (gfx1250) — matrix operand order repaired and checked by reading, not compiled, not run on hardware
//
#include <hip/hip_runtime.h>
#include <hip/hip_bf16.h>
#include <math.h>

// ---------------- CDNA5 WMMA types ----------------
typedef __attribute__((ext_vector_type(16))) _Float16 v16h;
typedef __attribute__((ext_vector_type(8)))  float    v8f;
typedef __attribute__((ext_vector_type(2)))  _Float16 h2;

// ---------------- problem constants ----------------
#define B_    2
#define N_    8192          // T*H*W tokens per batch
#define BTOK  16384         // B_*N_
#define C_    512
#define FF_   1024
#define GH_   8             // global (performer) heads
#define DH_   32            // head dim
#define NBF   110           // performer features
#define WIN_  64
#define NW_   128           // N_/WIN_

#define DN_CONST      0.4204482076268573f    // 32^-0.25
#define HALF_DN2      0.0883883476483184f    // 0.5 * 32^-0.5
#define MSQ_CONST     0.09534625892455922f   // 110^-0.5
#define LSCALE        0.17677669529663687f   // 32^-0.5
#define LN10000       9.210340371976184f

// =====================================================================
// Async global->LDS copy (CDNA5): 16 bytes per lane, tracked on ASYNCcnt
// =====================================================================
__device__ __forceinline__ void async_copy16(const _Float16* g, _Float16* l) {
  unsigned loff = (unsigned)(unsigned long long)(void*)l;  // low 32 bits = LDS byte addr
  asm volatile("global_load_async_to_lds_b128 %0, %1, off"
               :: "v"(loff), "v"(g) : "memory");
}
__device__ __forceinline__ void wait_async0() {
  asm volatile("s_wait_asynccnt 0" ::: "memory");
}

// =====================================================================
// WMMA GEMM: Cout[M,N] (= or +=) A[M,K] * WT[N,K]^T + bias, opt GELU
// A, WT are f16 in global; LDS double-buffered via async-to-LDS b128.
// Block: 256 thr = 8 waves; tile BM=128 x BN=64; each wave 32x32.
// mode: bit0 accumulate into CoutF, bit1 gelu, bit2 write f16 to CoutH
// =====================================================================
#define BM 128
#define BN 64
#define BK 32
#define ASTR 40   // padded stride (halves): rows 80B apart, 16B aligned chunks
#define BSTR 40

__global__ __launch_bounds__(256)
void wmma_gemm_kernel(const _Float16* __restrict__ A, const _Float16* __restrict__ WT,
                      const float* __restrict__ bias, float* __restrict__ CoutF,
                      _Float16* __restrict__ CoutH, int M, int N, int K, int mode) {
  __shared__ _Float16 sA[2][BM * ASTR];
  __shared__ _Float16 sB[2][BN * BSTR];
  const int tid  = threadIdx.x;
  const int lane = tid & 31;
  const int wave = tid >> 5;
  const int wm   = wave >> 1;    // 0..3
  const int wn   = wave & 1;     // 0..1
  const int lr   = lane & 15;
  const int half = lane >> 4;
  const int m0   = blockIdx.x * BM;
  const int n0   = blockIdx.y * BN;

  // per-thread staging coordinates (A: 2 chunks of 8 halves, B: 1 chunk)
  const int ar0 = tid >> 2,           ac0 = (tid & 3) * 8;
  const int ar1 = (tid + 256) >> 2,   ac1 = ((tid + 256) & 3) * 8;
  const int bn_ = tid >> 2,           bk_ = (tid & 3) * 8;

  v8f zero = {};
  v8f acc[2][2];
  acc[0][0] = zero; acc[0][1] = zero; acc[1][0] = zero; acc[1][1] = zero;

  // prologue: stage k0 = 0 into buffer 0
  {
    const _Float16* Ab = A + (size_t)m0 * K;
    async_copy16(Ab + (size_t)ar0 * K + ac0, &sA[0][ar0 * ASTR + ac0]);
    async_copy16(Ab + (size_t)ar1 * K + ac1, &sA[0][ar1 * ASTR + ac1]);
    async_copy16(WT + (size_t)(n0 + bn_) * K + bk_, &sB[0][bn_ * BSTR + bk_]);
  }

  int cur = 0;
  for (int k0 = 0; k0 < K; k0 += BK) {
    wait_async0();          // own async copies for 'cur' complete
    __syncthreads();        // everyone's copies visible; prior readers done
    const int nb = cur ^ 1;
    if (k0 + BK < K) {      // stage next tile into the other buffer
      const _Float16* Ab = A + (size_t)m0 * K + (k0 + BK);
      async_copy16(Ab + (size_t)ar0 * K + ac0, &sA[nb][ar0 * ASTR + ac0]);
      async_copy16(Ab + (size_t)ar1 * K + ac1, &sA[nb][ar1 * ASTR + ac1]);
      async_copy16(WT + (size_t)(n0 + bn_) * K + (k0 + BK) + bk_,
                   &sB[nb][bn_ * BSTR + bk_]);
    }

    const _Float16* sAc = sA[cur];
    const _Float16* sBc = sB[cur];
    // A fragments: 16-bit A 16x32 layout (lanes 0-15 M rows, halves split K)
    v16h afrag[2], bfrag[2];
#pragma unroll
    for (int mi = 0; mi < 2; ++mi) {
      int row = wm * 32 + mi * 16 + lr;
#pragma unroll
      for (int vv = 0; vv < 8; ++vv) {
        int kb = ((vv < 4) ? (2 * vv) : (16 + 2 * (vv - 4))) + 8 * half;
        h2 p = *(const h2*)&sAc[row * ASTR + kb];
        afrag[mi][2 * vv]     = p.x;
        afrag[mi][2 * vv + 1] = p.y;
      }
    }
    // B fragments: lanes are N cols; same K order as A fragments
#pragma unroll
    for (int ni = 0; ni < 2; ++ni) {
      int col = wn * 32 + ni * 16 + lr;
#pragma unroll
      for (int vv = 0; vv < 8; ++vv) {
        int kb = ((vv < 4) ? (2 * vv) : (16 + 2 * (vv - 4))) + 8 * half;
        h2 p = *(const h2*)&sBc[col * BSTR + kb];
        bfrag[ni][2 * vv]     = p.x;
        bfrag[ni][2 * vv + 1] = p.y;
      }
    }
#pragma unroll
    for (int mi = 0; mi < 2; ++mi)
#pragma unroll
      for (int ni = 0; ni < 2; ++ni)
        acc[mi][ni] = __builtin_amdgcn_wmma_f32_16x16x32_f16(
            false, afrag[mi], false, bfrag[ni], (short)0, acc[mi][ni], false, false);
    cur = nb;
  }

  // writeback: C/D layout: lane -> N=lr, VGPR r -> M = r + 8*half
#pragma unroll
  for (int mi = 0; mi < 2; ++mi) {
#pragma unroll
    for (int ni = 0; ni < 2; ++ni) {
      int col = n0 + wn * 32 + ni * 16 + lr;
      float bv = bias[col];
#pragma unroll
      for (int r = 0; r < 8; ++r) {
        int row = m0 + wm * 32 + mi * 16 + half * 8 + r;
        float val = acc[mi][ni][r] + bv;
        if (mode & 2)
          val = 0.5f * val * (1.0f + erff(val * 0.7071067811865475f));
        size_t off = (size_t)row * N + col;
        if (mode & 4) {
          CoutH[off] = (_Float16)val;
        } else {
          if (mode & 1) val += CoutF[off];
          CoutF[off] = val;
        }
      }
    }
  }
}

// =====================================================================
// Weight convert+transpose: WT[n*K+k] = (f16) W[k*N+n]
// =====================================================================
__global__ __launch_bounds__(256)
void cvtT_kernel(const float* __restrict__ W, _Float16* __restrict__ WT,
                 int K, int N) {
  int idx = blockIdx.x * 256 + threadIdx.x;
  int k = idx / N, n = idx - k * N;
  WT[(size_t)n * K + k] = (_Float16)W[idx];
}

// =====================================================================
// LayerNorm over C=512, one block per token; emits f16 activations
// =====================================================================
__global__ __launch_bounds__(256)
void ln_kernel(const float* __restrict__ x, const float* __restrict__ g,
               const float* __restrict__ beta, _Float16* __restrict__ y) {
  __shared__ float red[256];
  const int row = blockIdx.x, tid = threadIdx.x;
  const float* xr = x + (size_t)row * C_;
  float v0 = xr[tid], v1 = xr[tid + 256];
  red[tid] = v0 + v1; __syncthreads();
  for (int o = 128; o > 0; o >>= 1) { if (tid < o) red[tid] += red[tid + o]; __syncthreads(); }
  float mean = red[0] * (1.0f / C_);
  __syncthreads();
  float d0 = v0 - mean, d1 = v1 - mean;
  red[tid] = d0 * d0 + d1 * d1; __syncthreads();
  for (int o = 128; o > 0; o >>= 1) { if (tid < o) red[tid] += red[tid + o]; __syncthreads(); }
  float rstd = rsqrtf(red[0] * (1.0f / C_) + 1e-5f);
  _Float16* yr = y + (size_t)row * C_;
  yr[tid]       = (_Float16)(d0 * rstd * g[tid]       + beta[tid]);
  yr[tid + 256] = (_Float16)(d1 * rstd * g[tid + 256] + beta[tid + 256]);
}

// =====================================================================
// Performer (FAVOR+) feature kernels
// =====================================================================
__device__ inline unsigned ford(float f) {
  unsigned u = __float_as_uint(f);
  return (u & 0x80000000u) ? ~u : (u | 0x80000000u);
}
__device__ inline float fback(unsigned u) {
  return (u & 0x80000000u) ? __uint_as_float(u ^ 0x80000000u) : __uint_as_float(~u);
}

__global__ void zero16_kernel(unsigned* p) { if (threadIdx.x < 16) p[threadIdx.x] = 0u; }

#define FEAT_ROWS 16

__global__ __launch_bounds__(128)
void feat_max_kernel(const float* __restrict__ kbuf, const float* __restrict__ proj,
                     unsigned* __restrict__ kstabU) {
  __shared__ float sproj[NBF * DH_];
  __shared__ float srow[DH_];
  __shared__ float sred[128];
  const int bh = blockIdx.y, tile = blockIdx.x, tid = threadIdx.x;
  const int b = bh >> 3, hh = bh & 7;
  for (int i = tid; i < NBF * DH_; i += 128) sproj[i] = proj[i];
  float mx = -1e30f;
  for (int r = 0; r < FEAT_ROWS; ++r) {
    int i = tile * FEAT_ROWS + r;
    __syncthreads();
    if (tid < DH_) srow[tid] = kbuf[((size_t)(b * N_ + i)) * C_ + hh * DH_ + tid];
    __syncthreads();
    if (tid < NBF) {
      float dd = 0.f;
#pragma unroll
      for (int d = 0; d < DH_; ++d) dd += srow[d] * sproj[tid * DH_ + d];
      mx = fmaxf(mx, dd * DN_CONST);
    }
  }
  sred[tid] = mx; __syncthreads();
  for (int o = 64; o > 0; o >>= 1) { if (tid < o) sred[tid] = fmaxf(sred[tid], sred[tid + o]); __syncthreads(); }
  if (tid == 0) atomicMax(&kstabU[bh], ford(sred[0]));
}

__global__ __launch_bounds__(128)
void feat_kernel(const float* __restrict__ xbuf, const float* __restrict__ proj,
                 float* __restrict__ feat, const unsigned* __restrict__ kstabU,
                 int is_query) {
  __shared__ float sproj[NBF * DH_];
  __shared__ float srow[DH_];
  __shared__ float sred[128];
  const int bh = blockIdx.y, tile = blockIdx.x, tid = threadIdx.x;
  const int b = bh >> 3, hh = bh & 7;
  for (int i = tid; i < NBF * DH_; i += 128) sproj[i] = proj[i];
  float gstab = 0.f;
  if (!is_query) gstab = fback(kstabU[bh]);
  for (int r = 0; r < FEAT_ROWS; ++r) {
    int i = tile * FEAT_ROWS + r;
    __syncthreads();
    if (tid < DH_) srow[tid] = xbuf[((size_t)(b * N_ + i)) * C_ + hh * DH_ + tid];
    __syncthreads();
    float dd = -1e30f;
    if (tid < NBF) {
      float a = 0.f;
#pragma unroll
      for (int d = 0; d < DH_; ++d) a += srow[d] * sproj[tid * DH_ + d];
      dd = a * DN_CONST;
    }
    float diag = 0.f;
#pragma unroll
    for (int d = 0; d < DH_; ++d) { float qv = srow[d]; diag += qv * qv; }
    diag *= HALF_DN2;
    float stab = gstab;
    if (is_query) {
      sred[tid] = dd; __syncthreads();
      for (int o = 64; o > 0; o >>= 1) { if (tid < o) sred[tid] = fmaxf(sred[tid], sred[tid + o]); __syncthreads(); }
      stab = sred[0];
      __syncthreads();
    }
    if (tid < NBF)
      feat[((size_t)bh * N_ + i) * NBF + tid] = MSQ_CONST * (expf(dd - diag - stab) + 1e-4f);
  }
}

__global__ __launch_bounds__(128)
void ksum_kernel(const float* __restrict__ kp, float* __restrict__ ksum) {
  int bh = blockIdx.x, m = threadIdx.x;
  if (m >= NBF) return;
  float s = 0.f;
  const float* base = kp + (size_t)bh * N_ * NBF + m;
  for (int i = 0; i < N_; ++i) s += base[(size_t)i * NBF];
  ksum[bh * NBF + m] = s;
}

__global__ __launch_bounds__(256)
void ctx_kernel(const float* __restrict__ kp, const float* __restrict__ v,
                float* __restrict__ ctx) {
  int bh = blockIdx.y, mt = blockIdx.x, tid = threadIdx.x;
  int m = mt * 8 + (tid >> 5), d = tid & 31;
  int b = bh >> 3, hh = bh & 7;
  if (m >= NBF) return;
  float s = 0.f;
  const float* kpb = kp + (size_t)bh * N_ * NBF + m;
  const float* vb  = v  + (size_t)b * N_ * C_ + hh * DH_ + d;
  for (int i = 0; i < N_; ++i) s += kpb[(size_t)i * NBF] * vb[(size_t)i * C_];
  ctx[((size_t)bh * NBF + m) * DH_ + d] = s;
}

__global__ __launch_bounds__(256)
void og_kernel(const float* __restrict__ qp, const float* __restrict__ ksum,
               const float* __restrict__ ctx, _Float16* __restrict__ o) {
  __shared__ float sctx[NBF * DH_];
  __shared__ float sks[NBF];
  int bh = blockIdx.y, rt = blockIdx.x, tid = threadIdx.x;
  int b = bh >> 3, hh = bh & 7;
  for (int i = tid; i < NBF * DH_; i += 256) sctx[i] = ctx[(size_t)bh * NBF * DH_ + i];
  if (tid < NBF) sks[tid] = ksum[bh * NBF + tid];
  __syncthreads();
  int r = tid >> 5, d = tid & 31;
  int i = rt * 8 + r;
  const float* qpr = qp + ((size_t)bh * N_ + i) * NBF;
  float s1 = 0.f, acc = 0.f;
  for (int m = 0; m < NBF; ++m) {
    float qv = qpr[m];
    s1  += qv * sks[m];
    acc += qv * sctx[m * DH_ + d];
  }
  o[((size_t)(b * N_ + i)) * C_ + hh * DH_ + d] = (_Float16)(acc / s1);
}

// =====================================================================
// RoPE (local heads, cols 256..511), in place on q and k
// =====================================================================
__global__ __launch_bounds__(128)
void rope_kernel(float* __restrict__ q, float* __restrict__ k) {
  int tok = blockIdx.x, tid = threadIdx.x;
  int pos = tok & (N_ - 1);
  int hh = tid >> 4, d = tid & 15;
  int c1 = C_ / 2 + hh * DH_ + d, c2 = c1 + 16;
  float freq = expf(-(2.0f * d / DH_) * LN10000);
  float ang = (float)pos * freq;
  float sn, cs;
  sincosf(ang, &sn, &cs);
  size_t off = (size_t)tok * C_;
  float x1 = q[off + c1], x2 = q[off + c2];
  q[off + c1] = x1 * cs - x2 * sn;
  q[off + c2] = x2 * cs + x1 * sn;
  x1 = k[off + c1]; x2 = k[off + c2];
  k[off + c1] = x1 * cs - x2 * sn;
  k[off + c2] = x2 * cs + x1 * sn;
}

// =====================================================================
// Local windowed attention (local heads), block = 1 window (64 thr)
// =====================================================================
__global__ __launch_bounds__(64)
void local_attn_kernel(const float* __restrict__ q, const float* __restrict__ k,
                       const float* __restrict__ v, _Float16* __restrict__ o) {
  __shared__ float skk[3 * WIN_ * DH_];
  __shared__ float svv[3 * WIN_ * DH_];
  const int w = blockIdx.x, lh = blockIdx.y, b = blockIdx.z;
  const int tid = threadIdx.x;
  const int colbase = C_ / 2 + lh * DH_;
  const int gk0 = w * WIN_ - WIN_;
  for (int idx = tid; idx < 3 * WIN_ * DH_; idx += 64) {
    int j = idx >> 5, d = idx & 31;
    int gk = gk0 + j;
    float kv = 0.f, vv = 0.f;
    if (gk >= 0 && gk < N_) {
      size_t ro = ((size_t)(b * N_ + gk)) * C_ + colbase;
      kv = k[ro + d]; vv = v[ro + d];
    }
    skk[idx] = kv; svv[idx] = vv;
  }
  __syncthreads();
  const int jlo = (gk0 < 0) ? -gk0 : 0;
  const int jhi = (gk0 + 3 * WIN_ > N_) ? (N_ - gk0) : (3 * WIN_);
  float qr[DH_];
  size_t qoff = ((size_t)(b * N_ + w * WIN_ + tid)) * C_ + colbase;
#pragma unroll
  for (int d = 0; d < DH_; ++d) qr[d] = q[qoff + d] * LSCALE;
  float mx = -1e30f;
  for (int j = jlo; j < jhi; ++j) {
    float s = 0.f;
#pragma unroll
    for (int d = 0; d < DH_; ++d) s += qr[d] * skk[j * DH_ + d];
    mx = fmaxf(mx, s);
  }
  float sum = 0.f;
  float oacc[DH_];
#pragma unroll
  for (int d = 0; d < DH_; ++d) oacc[d] = 0.f;
  for (int j = jlo; j < jhi; ++j) {
    float s = 0.f;
#pragma unroll
    for (int d = 0; d < DH_; ++d) s += qr[d] * skk[j * DH_ + d];
    float e = expf(s - mx);
    sum += e;
#pragma unroll
    for (int d = 0; d < DH_; ++d) oacc[d] += e * svv[j * DH_ + d];
  }
  float inv = 1.0f / sum;
#pragma unroll
  for (int d = 0; d < DH_; ++d) o[qoff + d] = (_Float16)(oacc[d] * inv);
}

// =====================================================================
// Host orchestration
// =====================================================================
extern "C" void kernel_launch(void* const* d_in, const int* in_sizes, int n_in,
                              void* d_out, int out_size, void* d_ws, size_t ws_size,
                              hipStream_t stream) {
  (void)in_sizes; (void)n_in; (void)out_size; (void)ws_size;
  const float* x    = (const float*)d_in[0];
  const float* ln1g = (const float*)d_in[1];
  const float* ln1b = (const float*)d_in[2];
  const float* Wq = (const float*)d_in[3];  const float* bq = (const float*)d_in[4];
  const float* Wk = (const float*)d_in[5];  const float* bk = (const float*)d_in[6];
  const float* Wv = (const float*)d_in[7];  const float* bv = (const float*)d_in[8];
  const float* Wo = (const float*)d_in[9];  const float* bo = (const float*)d_in[10];
  const float* proj = (const float*)d_in[11];
  const float* ln2g = (const float*)d_in[12]; const float* ln2b = (const float*)d_in[13];
  const float* W1 = (const float*)d_in[14]; const float* b1 = (const float*)d_in[15];
  const float* W2 = (const float*)d_in[16]; const float* b2 = (const float*)d_in[17];

  char* ws = (char*)d_ws;
  const size_t SZ   = (size_t)BTOK * C_ * sizeof(float);           // 33.5 MB
  const size_t HSZ  = (size_t)BTOK * C_ * sizeof(_Float16);        // 16.8 MB
  const size_t FEAT = (size_t)B_ * GH_ * N_ * NBF * sizeof(float); // 57.7 MB

  float* h  = (float*)(ws);
  float* q  = (float*)(ws + SZ);
  float* k  = (float*)(ws + 2 * SZ);
  float* v  = (float*)(ws + 3 * SZ);
  float* qp = (float*)(ws + 4 * SZ);
  float* kp = (float*)(ws + 4 * SZ + FEAT);
  _Float16* ff16  = (_Float16*)qp;               // reuse qp/kp region after attention
  _Float16* act16 = (_Float16*)(ws + 4 * SZ + 2 * FEAT);
  _Float16* wT    = (_Float16*)(ws + 4 * SZ + 2 * FEAT + HSZ);     // 4M f16 = 8.4 MB
  char* tail = ws + 4 * SZ + 2 * FEAT + HSZ + (size_t)4194304 * sizeof(_Float16);
  float*    ctx    = (float*)tail;
  float*    ksum   = (float*)(tail + (size_t)16 * NBF * DH_ * 4);
  unsigned* kstabU = (unsigned*)(tail + (size_t)16 * NBF * DH_ * 4 + (size_t)16 * NBF * 4);

  // f16 transposed weight blob layout (per layer stride = 2097152 elems)
  const size_t LW = 2097152;
  const size_t OQ = 0, OK = 262144, OV = 524288, OO = 786432, O1 = 1048576, O2 = 1572864;

  // ---- one-time weight conversion (deterministic every call) ----
  for (int l = 0; l < 2; ++l) {
    cvtT_kernel<<<(C_ * C_) / 256, 256, 0, stream>>>(Wq + (size_t)l * C_ * C_,  wT + l * LW + OQ, C_, C_);
    cvtT_kernel<<<(C_ * C_) / 256, 256, 0, stream>>>(Wk + (size_t)l * C_ * C_,  wT + l * LW + OK, C_, C_);
    cvtT_kernel<<<(C_ * C_) / 256, 256, 0, stream>>>(Wv + (size_t)l * C_ * C_,  wT + l * LW + OV, C_, C_);
    cvtT_kernel<<<(C_ * C_) / 256, 256, 0, stream>>>(Wo + (size_t)l * C_ * C_,  wT + l * LW + OO, C_, C_);
    cvtT_kernel<<<(C_ * FF_) / 256, 256, 0, stream>>>(W1 + (size_t)l * C_ * FF_, wT + l * LW + O1, C_, FF_);
    cvtT_kernel<<<(FF_ * C_) / 256, 256, 0, stream>>>(W2 + (size_t)l * FF_ * C_, wT + l * LW + O2, FF_, C_);
  }

  hipMemcpyAsync(h, x, SZ, hipMemcpyDeviceToDevice, stream);

  for (int l = 0; l < 2; ++l) {
    const float* pj = proj + (size_t)l * NBF * DH_;
    const _Float16* wl = wT + (size_t)l * LW;

    ln_kernel<<<BTOK, 256, 0, stream>>>(h, ln1g + l * C_, ln1b + l * C_, act16);
    wmma_gemm_kernel<<<dim3(BTOK / BM, C_ / BN), 256, 0, stream>>>(
        act16, wl + OQ, bq + l * C_, q, nullptr, BTOK, C_, C_, 0);
    wmma_gemm_kernel<<<dim3(BTOK / BM, C_ / BN), 256, 0, stream>>>(
        act16, wl + OK, bk + l * C_, k, nullptr, BTOK, C_, C_, 0);
    wmma_gemm_kernel<<<dim3(BTOK / BM, C_ / BN), 256, 0, stream>>>(
        act16, wl + OV, bv + l * C_, v, nullptr, BTOK, C_, C_, 0);

    // Performer global heads
    zero16_kernel<<<1, 32, 0, stream>>>(kstabU);
    feat_max_kernel<<<dim3(N_ / FEAT_ROWS, 16), 128, 0, stream>>>(k, pj, kstabU);
    feat_kernel<<<dim3(N_ / FEAT_ROWS, 16), 128, 0, stream>>>(q, pj, qp, kstabU, 1);
    feat_kernel<<<dim3(N_ / FEAT_ROWS, 16), 128, 0, stream>>>(k, pj, kp, kstabU, 0);
    ksum_kernel<<<16, 128, 0, stream>>>(kp, ksum);
    ctx_kernel<<<dim3((NBF + 7) / 8, 16), 256, 0, stream>>>(kp, v, ctx);
    og_kernel<<<dim3(N_ / 8, 16), 256, 0, stream>>>(qp, ksum, ctx, act16);

    // Local heads: RoPE + windowed attention
    rope_kernel<<<BTOK, 128, 0, stream>>>(q, k);
    local_attn_kernel<<<dim3(NW_, GH_, B_), 64, 0, stream>>>(q, k, v, act16);

    // h += o @ Wo + bo
    wmma_gemm_kernel<<<dim3(BTOK / BM, C_ / BN), 256, 0, stream>>>(
        act16, wl + OO, bo + l * C_, h, nullptr, BTOK, C_, C_, 1);

    // FFN
    ln_kernel<<<BTOK, 256, 0, stream>>>(h, ln2g + l * C_, ln2b + l * C_, act16);
    wmma_gemm_kernel<<<dim3(BTOK / BM, FF_ / BN), 256, 0, stream>>>(
        act16, wl + O1, b1 + l * FF_, nullptr, ff16, BTOK, FF_, C_, 2 | 4);
    wmma_gemm_kernel<<<dim3(BTOK / BM, C_ / BN), 256, 0, stream>>>(
        ff16, wl + O2, b2 + l * C_, h, nullptr, BTOK, C_, FF_, 1);
  }

  hipMemcpyAsync(d_out, h, SZ, hipMemcpyDeviceToDevice, stream);
}
